// GraphTransformer_37563783971456
// MI455X (gfx1250) — compile-verified, matrix operand order checked by reading, not run on hardware
//
#include <hip/hip_runtime.h>
#include <hip/hip_bf16.h>
#include <math.h>

// ---------------- problem constants (match reference) ----------------
#define NNODES 16384
#define NEDGES 262144
#define DIN    384
#define DHID   256
#define NHEAD  8
#define CHEAD  32
#define NLAYER 3
#define LN_EPS 1e-5f

typedef __attribute__((ext_vector_type(16))) __bf16 v16bf;
typedef __attribute__((ext_vector_type(8)))  float  v8f;
typedef unsigned short u16;
typedef unsigned int   u32;
typedef __attribute__((ext_vector_type(8)))  u16 u16x8;
typedef __attribute__((ext_vector_type(16))) u16 u16x16;

// ---------------- fp32 -> bf16 (round to nearest even) ----------------
__device__ __forceinline__ u16 f32_to_bf16_rne(float f) {
  u32 u = __float_as_uint(f);
  u32 r = u + 0x7FFFu + ((u >> 16) & 1u);
  return (u16)(r >> 16);
}

// linear convert (activations, row-major [M,K] stays row-major)
__global__ void cvt_bf16_kernel(const float* __restrict__ in, u16* __restrict__ out, int n) {
  int i = blockIdx.x * blockDim.x + threadIdx.x;
  if (i < n) out[i] = f32_to_bf16_rne(in[i]);
}

// transpose convert (weights): in W[K,Nc] f32 -> out Wt[Nc,K] bf16
__global__ void cvt_bf16_t_kernel(const float* __restrict__ in, u16* __restrict__ out,
                                  int K, int Nc) {
  int idx = blockIdx.x * blockDim.x + threadIdx.x;
  if (idx >= K * Nc) return;
  int n = idx / K, k = idx - n * K;
  out[idx] = f32_to_bf16_rne(in[(size_t)k * Nc + n]);
}

__device__ __forceinline__ v16bf frag_cat(u16x8 lo, u16x8 hi) {
  u16x16 f = __builtin_shufflevector(lo, hi, 0,1,2,3,4,5,6,7,8,9,10,11,12,13,14,15);
  v16bf r;
  __builtin_memcpy(&r, &f, sizeof(r));
  return r;
}

// ---------------- WMMA GEMM: C[M,Nc] = A[M,K](bf16) @ Wt[Nc,K](bf16)^T + bias, optional GELU ----
// Register-blocked: one wave computes a 32x32 output block = 2x2 tiles of 16x16.
// Per K-step (32): 2 A fragments + 2 B fragments (8x global_load_b128) feed 4 WMMAs
// -> 2 loads per v_wmma_f32_16x16x32_bf16 (double the arithmetic intensity of 1 tile/wave).
// Fragment layouts per CDNA5 ISA 7.12.2 (wave32):
//   A 16x32 bf16: lanes 0-15 row M=lane hold K 0..7 / 16..23; lanes 16-31 hold K 8..15 / 24..31.
//   B 32x16 bf16: mirrored with N in place of M (Wt[Nc,K] makes both sides K-contiguous).
//   C/D 16x16 f32: acc[i] -> row m + i + 8*(lane>=16), col n + (lane&15).
__global__ __launch_bounds__(256)
void wmma_gemm_bias_kernel(const u16* __restrict__ A, const u16* __restrict__ Wt,
                           const float* __restrict__ bias, float* __restrict__ Cout,
                           int M, int K, int Nc, int gelu) {
  int wave = (int)((blockIdx.x * blockDim.x + threadIdx.x) >> 5);
  int lane = threadIdx.x & 31;
  int blk_n = Nc >> 5;                       // 32-wide N blocks
  int blk_total = (M >> 5) * blk_n;
  if (wave >= blk_total) return;
  int bm = wave / blk_n, bn = wave % blk_n;
  int m0 = bm << 5, n0 = bn << 5;
  int half = lane >> 4;                      // 0 or 1
  int sub  = lane & 15;
  int kgrp = half << 3;                      // 0 or 8

  // 32-bit element offsets from kernel-arg bases (saddr + voffset addressing)
  int a_off = (m0 + sub) * K + kgrp;         // 16B aligned (K multiple of 16)
  int b_off = (n0 + sub) * K + kgrp;
  int aK = 16 * K;                           // stride to second M/N tile

  v8f acc00 = {0.f,0.f,0.f,0.f,0.f,0.f,0.f,0.f};
  v8f acc01 = acc00, acc10 = acc00, acc11 = acc00;

  for (int k0 = 0; k0 < K; k0 += 32) {
    v16bf a0 = frag_cat(*(const u16x8*)(A + a_off + k0),
                        *(const u16x8*)(A + a_off + k0 + 16));
    v16bf a1 = frag_cat(*(const u16x8*)(A + a_off + aK + k0),
                        *(const u16x8*)(A + a_off + aK + k0 + 16));
    v16bf b0 = frag_cat(*(const u16x8*)(Wt + b_off + k0),
                        *(const u16x8*)(Wt + b_off + k0 + 16));
    v16bf b1 = frag_cat(*(const u16x8*)(Wt + b_off + aK + k0),
                        *(const u16x8*)(Wt + b_off + aK + k0 + 16));
    acc00 = __builtin_amdgcn_wmma_f32_16x16x32_bf16(false, a0, false, b0, (short)0, acc00, false, false);
    acc01 = __builtin_amdgcn_wmma_f32_16x16x32_bf16(false, a0, false, b1, (short)0, acc01, false, false);
    acc10 = __builtin_amdgcn_wmma_f32_16x16x32_bf16(false, a1, false, b0, (short)0, acc10, false, false);
    acc11 = __builtin_amdgcn_wmma_f32_16x16x32_bf16(false, a1, false, b1, (short)0, acc11, false, false);
  }

  // epilogue: bias + optional exact GELU, store 4 tiles
  int crow = (half << 3);
#pragma unroll
  for (int t = 0; t < 4; ++t) {
    v8f acc = (t == 0) ? acc00 : (t == 1) ? acc01 : (t == 2) ? acc10 : acc11;
    int mt = m0 + ((t >> 1) << 4) + crow;
    int ccol = n0 + ((t & 1) << 4) + sub;
    float bb = bias ? bias[ccol] : 0.f;
#pragma unroll
    for (int i = 0; i < 8; ++i) {
      float v = acc[i] + bb;
      if (gelu) v = 0.5f * v * (1.0f + erff(v * 0.70710678118654752f));
      Cout[(size_t)(mt + i) * Nc + ccol] = v;
    }
  }
}

// ---------------- attention edge-phase kernels ----------------
__global__ void attn_init_kernel(float* __restrict__ m, float* __restrict__ den,
                                 float* __restrict__ attn, int nh, int ntot) {
  int i = blockIdx.x * blockDim.x + threadIdx.x;
  if (i < ntot) attn[i] = 0.f;
  if (i < nh) { m[i] = -__builtin_inff(); den[i] = 0.f; }
}

__global__ void edge_score_kernel(const float* __restrict__ q, const float* __restrict__ k,
                                  const int* __restrict__ src, const int* __restrict__ dst,
                                  float* __restrict__ score) {
  int idx = blockIdx.x * blockDim.x + threadIdx.x;
  if (idx >= NEDGES * NHEAD) return;
  int e = idx >> 3, h = idx & 7;
  const float4* qr = (const float4*)(q + (size_t)dst[e] * DHID + h * CHEAD);
  const float4* kr = (const float4*)(k + (size_t)src[e] * DHID + h * CHEAD);
  float acc = 0.f;
#pragma unroll
  for (int j = 0; j < 8; ++j) {
    float4 a = qr[j], b = kr[j];
    acc += a.x * b.x + a.y * b.y + a.z * b.z + a.w * b.w;
  }
  score[idx] = acc * 0.17677669529663689f;   // 1/sqrt(32)
}

__device__ __forceinline__ void atomicMaxF(float* addr, float v) {
  // ordered-int trick: ints compare like floats for >=0, reversed (as unsigned) for <0
  if (v >= 0.f) atomicMax((int*)addr, __float_as_int(v));
  else          atomicMin((u32*)addr, (u32)__float_as_int(v));
}

__global__ void edge_max_kernel(const float* __restrict__ score, const int* __restrict__ dst,
                                float* __restrict__ m) {
  int idx = blockIdx.x * blockDim.x + threadIdx.x;
  if (idx >= NEDGES * NHEAD) return;
  int e = idx >> 3, h = idx & 7;
  atomicMaxF(&m[(size_t)dst[e] * NHEAD + h], score[idx]);
}

__global__ void edge_exp_kernel(float* __restrict__ score, const int* __restrict__ dst,
                                const float* __restrict__ m, float* __restrict__ den) {
  int idx = blockIdx.x * blockDim.x + threadIdx.x;
  if (idx >= NEDGES * NHEAD) return;
  int e = idx >> 3, h = idx & 7;
  float ex = expf(score[idx] - m[(size_t)dst[e] * NHEAD + h]);
  score[idx] = ex;                                   // alpha numerator, reuse buffer
  atomicAdd(&den[(size_t)dst[e] * NHEAD + h], ex);
}

__global__ void edge_scatter_kernel(const float* __restrict__ score, const float* __restrict__ den,
                                    const float* __restrict__ v, const int* __restrict__ src,
                                    const int* __restrict__ dst, float* __restrict__ attn) {
  int idx = blockIdx.x * blockDim.x + threadIdx.x;
  if (idx >= NEDGES * NHEAD) return;
  int e = idx >> 3, h = idx & 7;
  int s = src[e], d = dst[e];
  float w = score[idx] / den[(size_t)d * NHEAD + h];
  const float* vr = v + (size_t)s * DHID + h * CHEAD;
  float* o = attn + (size_t)d * DHID + h * CHEAD;
#pragma unroll
  for (int c = 0; c < CHEAD; ++c) atomicAdd(o + c, w * vr[c]);   // global_atomic_add_f32, L2-resident
}

// ---------------- fused residual-add + LayerNorm (one wave per 256-wide row) ----------------
__global__ __launch_bounds__(256)
void add_ln_kernel(const float* __restrict__ a, const float* __restrict__ b,
                   const float* __restrict__ c, const float* __restrict__ g,
                   const float* __restrict__ beta, float* __restrict__ out, int rows) {
  int wave = (int)((blockIdx.x * blockDim.x + threadIdx.x) >> 5);
  int lane = threadIdx.x & 31;
  if (wave >= rows) return;
  size_t base = (size_t)wave * DHID;
  float vals[8];
  float sum = 0.f;
#pragma unroll
  for (int j = 0; j < 8; ++j) {
    int col = lane + (j << 5);                   // coalesced across lanes
    float v = a[base + col] + b[base + col];
    if (c) v += c[base + col];
    vals[j] = v; sum += v;
  }
#pragma unroll
  for (int off = 16; off > 0; off >>= 1) sum += __shfl_xor(sum, off, 32);
  float mu = sum * (1.f / DHID);
  float var = 0.f;
#pragma unroll
  for (int j = 0; j < 8; ++j) { float d = vals[j] - mu; var += d * d; }
#pragma unroll
  for (int off = 16; off > 0; off >>= 1) var += __shfl_xor(var, off, 32);
  float rs = rsqrtf(var * (1.f / DHID) + LN_EPS);
#pragma unroll
  for (int j = 0; j < 8; ++j) {
    int col = lane + (j << 5);
    out[base + col] = (vals[j] - mu) * rs * g[col] + beta[col];
  }
}

// ---------------- host orchestration ----------------
extern "C" void kernel_launch(void* const* d_in, const int* in_sizes, int n_in,
                              void* d_out, int out_size, void* d_ws, size_t ws_size,
                              hipStream_t stream) {
  (void)in_sizes; (void)n_in; (void)out_size; (void)ws_size;

  // Input flattening (setup_inputs dict order, params depth-first):
  // 0:x  1:edge_index
  // per layer i (base 2+10*i): Wq bq Wk bk Wv bv Ws bs ln_g ln_b
  // 32..35: W1 b1 W2 b2     36,37: Wo bo
  const float* x  = (const float*)d_in[0];
  const int*   ei = (const int*)d_in[1];
  const int*   src = ei;
  const int*   dst = ei + NEDGES;
  const float* W1 = (const float*)d_in[32];
  const float* b1 = (const float*)d_in[33];
  const float* W2 = (const float*)d_in[34];
  const float* b2 = (const float*)d_in[35];
  const float* Wo = (const float*)d_in[36];
  const float* bo = (const float*)d_in[37];

  // workspace carve (256B aligned)
  char* wp = (char*)d_ws;
  auto carve = [&](size_t bytes) { void* p = (void*)wp; wp += (bytes + 255) & ~(size_t)255; return p; };
  float* qkvs = (float*)carve((size_t)NNODES * 1024 * sizeof(float));  // q|k|v|s, reused as FFN hidden
  float* qb = qkvs;
  float* kb = qkvs + (size_t)NNODES * DHID;
  float* vb = qkvs + (size_t)NNODES * DHID * 2;
  float* sb = qkvs + (size_t)NNODES * DHID * 3;
  float* bufA = (float*)carve((size_t)NNODES * DHID * sizeof(float));  // attn out / ffn out
  float* bufH = (float*)carve((size_t)NNODES * DHID * sizeof(float));  // layer output h
  float* bufT = (float*)carve((size_t)NNODES * DHID * sizeof(float));  // post-LN1 (ffn input + residual)
  float* mbuf = (float*)carve((size_t)NNODES * NHEAD * sizeof(float));
  float* dbuf = (float*)carve((size_t)NNODES * NHEAD * sizeof(float));
  float* score = (float*)carve((size_t)NEDGES * NHEAD * sizeof(float));
  u16* actbf = (u16*)carve((size_t)NNODES * 1024 * sizeof(u16));
  u16* wtbf  = (u16*)carve((size_t)1024 * 256 * sizeof(u16));          // largest weight: 262144 elems

  auto cvt = [&](const float* in, u16* out, size_t n) {
    int blocks = (int)((n + 255) / 256);
    cvt_bf16_kernel<<<blocks, 256, 0, stream>>>(in, out, (int)n);
  };
  auto cvtT = [&](const float* in, u16* out, int K, int Nc) {   // W[K,Nc] -> Wt[Nc,K] bf16
    int n = K * Nc;
    int blocks = (n + 255) / 256;
    cvt_bf16_t_kernel<<<blocks, 256, 0, stream>>>(in, out, K, Nc);
  };
  auto gemm = [&](const u16* A, const u16* Wt, const float* bias, float* C,
                  int M, int K, int Nc, int gelu) {
    int waves = (M >> 5) * (Nc >> 5);            // one wave per 32x32 block
    int blocks = (waves + 7) / 8;                // 8 waves (256 threads) per block
    wmma_gemm_bias_kernel<<<blocks, 256, 0, stream>>>(A, Wt, bias, C, M, K, Nc, gelu);
  };

  const int EH = NEDGES * NHEAD;
  const int eh_blocks = (EH + 255) / 256;
  const int ln_blocks = NNODES / 8;              // one wave per row, 8 waves per block

  const float* hin = x;
  for (int l = 0; l < NLAYER; ++l) {
    int d = (l == 0) ? DIN : DHID;
    const float* Wq = (const float*)d_in[2 + 10 * l + 0];
    const float* bq = (const float*)d_in[2 + 10 * l + 1];
    const float* Wk = (const float*)d_in[2 + 10 * l + 2];
    const float* bk = (const float*)d_in[2 + 10 * l + 3];
    const float* Wv = (const float*)d_in[2 + 10 * l + 4];
    const float* bv = (const float*)d_in[2 + 10 * l + 5];
    const float* Ws = (const float*)d_in[2 + 10 * l + 6];
    const float* bs = (const float*)d_in[2 + 10 * l + 7];
    const float* lng = (const float*)d_in[2 + 10 * l + 8];
    const float* lnb = (const float*)d_in[2 + 10 * l + 9];

    // QKVS projections (stream-serialized, wtbf reused safely)
    cvt(hin, actbf, (size_t)NNODES * d);
    cvtT(Wq, wtbf, d, DHID);  gemm(actbf, wtbf, bq, qb, NNODES, d, DHID, 0);
    cvtT(Wk, wtbf, d, DHID);  gemm(actbf, wtbf, bk, kb, NNODES, d, DHID, 0);
    cvtT(Wv, wtbf, d, DHID);  gemm(actbf, wtbf, bv, vb, NNODES, d, DHID, 0);
    cvtT(Ws, wtbf, d, DHID);  gemm(actbf, wtbf, bs, sb, NNODES, d, DHID, 0);

    // edge attention with scatter-softmax
    int nh = NNODES * NHEAD, ntot = NNODES * DHID;
    attn_init_kernel<<<(ntot + 255) / 256, 256, 0, stream>>>(mbuf, dbuf, bufA, nh, ntot);
    edge_score_kernel<<<eh_blocks, 256, 0, stream>>>(qb, kb, src, dst, score);
    edge_max_kernel<<<eh_blocks, 256, 0, stream>>>(score, dst, mbuf);
    edge_exp_kernel<<<eh_blocks, 256, 0, stream>>>(score, dst, mbuf, dbuf);
    edge_scatter_kernel<<<eh_blocks, 256, 0, stream>>>(score, dbuf, vb, src, dst, bufA);

    // h1 = LN(conv_out + skip + residual)
    add_ln_kernel<<<ln_blocks, 256, 0, stream>>>(bufA, sb, (l > 0) ? hin : nullptr,
                                                 lng, lnb, bufT, NNODES);

    // FFN: gelu(h1 @ W1 + b1) @ W2 + b2  (hidden reuses qkvs buffer)
    cvt(bufT, actbf, (size_t)NNODES * DHID);
    cvtT(W1, wtbf, DHID, 1024);
    gemm(actbf, wtbf, b1, qkvs, NNODES, DHID, 1024, 1);
    cvt(qkvs, actbf, (size_t)NNODES * 1024);
    cvtT(W2, wtbf, 1024, DHID);
    gemm(actbf, wtbf, b2, bufA, NNODES, 1024, DHID, 0);

    // h = LN(h1 + f)
    add_ln_kernel<<<ln_blocks, 256, 0, stream>>>(bufT, bufA, nullptr, lng, lnb, bufH, NNODES);
    hin = bufH;
  }

  // output projection -> d_out (fp32, N x IN)
  cvt(hin, actbf, (size_t)NNODES * DHID);
  cvtT(Wo, wtbf, DHID, DIN);
  gemm(actbf, wtbf, bo, (float*)d_out, NNODES, DHID, DIN, 0);
}
